// MultiHeadAttentionEinops_1288490189528
// MI455X (gfx1250) — compile-verified, matrix operand order checked by reading, not run on hardware
//
#include <hip/hip_runtime.h>

typedef __attribute__((ext_vector_type(16))) _Float16 v16h;
typedef __attribute__((ext_vector_type(8)))  _Float16 v8h;
typedef __attribute__((ext_vector_type(8)))  float    v8f;

#define BB 16
#define NN 1024
#define DD 768
#define HH 12
#define EE 64
#define BN (BB * NN)

// ---------------------------------------------------------------------------
// Fragment loader: 16-bit A/B fragment for v_wmma_*_16x16x32_f16.
// ISA layout (CDNA5 ISA): lane&15 = row (A: M, B: N), hi = lane>>4 selects the
// K-offset: VGPRs 0-3 hold K = 8*hi + 0..7, VGPRs 4-7 hold K = 16 + 8*hi + 0..7.
// Both 8-half groups are contiguous -> two global_load_b128 per lane.
// ---------------------------------------------------------------------------
__device__ __forceinline__ v16h load_frag16(const _Float16* p0, int stride, int lane) {
    const int m   = lane & 15;
    const int off = (lane >> 4) << 3;
    const _Float16* p = p0 + (size_t)m * stride + off;
    v8h lo = *(const v8h*)(p);
    v8h hi = *(const v8h*)(p + 16);
    return __builtin_shufflevector(lo, hi, 0, 1, 2, 3, 4, 5, 6, 7,
                                           8, 9, 10, 11, 12, 13, 14, 15);
}

__device__ __forceinline__ v8f wmma_f16(v16h a, v16h b, v8f c) {
    return __builtin_amdgcn_wmma_f32_16x16x32_f16(false, a, false, b, (short)0, c,
                                                  false, false);
}

// ---------------------------------------------------------------------------
// f32 -> f16 conversion (grid-stride)
// ---------------------------------------------------------------------------
__global__ void cvt_f32_f16(const float* __restrict__ src, _Float16* __restrict__ dst, int n) {
    int i = blockIdx.x * blockDim.x + threadIdx.x;
    const int stride = gridDim.x * blockDim.x;
    for (; i < n; i += stride) dst[i] = (_Float16)src[i];
}

// ---------------------------------------------------------------------------
// One wave computes a 32x64 tile of C = A(32xK) * W(64 rows x K)^T.
// Software-pipelined: iteration k's 8 WMMAs run while iteration k+1's 12
// fragment loads are in flight (no s_wait_loadcnt 0 in front of the WMMAs).
// acc[0..3] = rows 0-15 x col tiles 0-3 ; acc[4..7] = rows 16-31.
// ---------------------------------------------------------------------------
__device__ __forceinline__ void gemm_32x64(const _Float16* __restrict__ A,
                                           const _Float16* __restrict__ W,
                                           int lane, v8f acc[8]) {
    constexpr int K = DD;
    v16h a0 = load_frag16(A + 0 * K, K, lane);
    v16h a1 = load_frag16(A + 16 * K, K, lane);
    v16h b0 = load_frag16(W + 0 * 16 * K, K, lane);
    v16h b1 = load_frag16(W + 1 * 16 * K, K, lane);
    v16h b2 = load_frag16(W + 2 * 16 * K, K, lane);
    v16h b3 = load_frag16(W + 3 * 16 * K, K, lane);
    for (int k = 0; k < K; k += 32) {
        const int kn = (k + 32 < K) ? (k + 32) : k;   // last iter: cheap re-fetch (cache hit)
        v16h na0 = load_frag16(A + kn, K, lane);
        v16h na1 = load_frag16(A + 16 * K + kn, K, lane);
        v16h nb0 = load_frag16(W + 0 * 16 * K + kn, K, lane);
        v16h nb1 = load_frag16(W + 1 * 16 * K + kn, K, lane);
        v16h nb2 = load_frag16(W + 2 * 16 * K + kn, K, lane);
        v16h nb3 = load_frag16(W + 3 * 16 * K + kn, K, lane);
        acc[0] = wmma_f16(a0, b0, acc[0]);
        acc[1] = wmma_f16(a0, b1, acc[1]);
        acc[2] = wmma_f16(a0, b2, acc[2]);
        acc[3] = wmma_f16(a0, b3, acc[3]);
        acc[4] = wmma_f16(a1, b0, acc[4]);
        acc[5] = wmma_f16(a1, b1, acc[5]);
        acc[6] = wmma_f16(a1, b2, acc[6]);
        acc[7] = wmma_f16(a1, b3, acc[7]);
        a0 = na0; a1 = na1;
        b0 = nb0; b1 = nb1; b2 = nb2; b3 = nb3;
    }
}

// ---------------------------------------------------------------------------
// QK projection: qk = x @ Wqk^T + b ; de-interleave feature f:
// h = f>>7, e = (f&127)>>1, sel = f&1 (0->Q, 1->K).  Q,K stored (b,h,n,64) f16.
// ---------------------------------------------------------------------------
__global__ void __launch_bounds__(128) proj_qk_kernel(
        const _Float16* __restrict__ xh, const _Float16* __restrict__ wqk,
        const float* __restrict__ bias,
        _Float16* __restrict__ qb, _Float16* __restrict__ kb) {
    const int lane = threadIdx.x & 31;
    const int wid  = blockIdx.x * (blockDim.x >> 5) + (threadIdx.x >> 5);
    const int nCG  = (2 * DD) / 64;          // 24 column groups
    const int rt   = wid / nCG;              // 32-row tile over B*N
    const int cg   = wid % nCG;
    if (rt >= BN / 32) return;

    v8f acc[8] = {};
    gemm_32x64(xh + (size_t)rt * 32 * DD, wqk + (size_t)cg * 64 * DD, lane, acc);

    const int n  = lane & 15;
    const int hi = lane >> 4;
#pragma unroll
    for (int t = 0; t < 4; ++t) {
        const int f  = cg * 64 + t * 16 + n;         // 0..1535
        const float bv = bias[f];
        const int h   = f >> 7;
        const int e   = (f & 127) >> 1;
        _Float16* dst = (f & 1) ? kb : qb;
#pragma unroll
        for (int sub = 0; sub < 2; ++sub) {
#pragma unroll
            for (int r = 0; r < 8; ++r) {
                const int g  = rt * 32 + sub * 16 + r + 8 * hi;   // row in B*N
                const int b  = g >> 10;
                const int nn = g & 1023;
                dst[(((size_t)b * HH + h) * NN + nn) * EE + e] =
                    (_Float16)(acc[sub * 4 + t][r] + bv);
            }
        }
    }
}

// ---------------------------------------------------------------------------
// V projection: v = x @ Wv^T + b, stored TRANSPOSED as Vt(b,h,e,n) f16 so the
// attn-output WMMA A-fragments are contiguous.
// ---------------------------------------------------------------------------
__global__ void __launch_bounds__(128) proj_v_kernel(
        const _Float16* __restrict__ xh, const _Float16* __restrict__ wv,
        const float* __restrict__ bias, _Float16* __restrict__ vt) {
    const int lane = threadIdx.x & 31;
    const int wid  = blockIdx.x * (blockDim.x >> 5) + (threadIdx.x >> 5);
    const int nCG  = DD / 64;                // 12
    const int rt   = wid / nCG;
    const int cg   = wid % nCG;
    if (rt >= BN / 32) return;

    v8f acc[8] = {};
    gemm_32x64(xh + (size_t)rt * 32 * DD, wv + (size_t)cg * 64 * DD, lane, acc);

    const int n  = lane & 15;
    const int hi = lane >> 4;
#pragma unroll
    for (int t = 0; t < 4; ++t) {
        const int f  = cg * 64 + t * 16 + n;         // 0..767
        const float bv = bias[f];
        const int h = f >> 6;
        const int e = f & 63;
#pragma unroll
        for (int sub = 0; sub < 2; ++sub) {
#pragma unroll
            for (int r = 0; r < 8; ++r) {
                const int g  = rt * 32 + sub * 16 + r + 8 * hi;
                const int b  = g >> 10;
                const int nn = g & 1023;
                vt[(((size_t)b * HH + h) * EE + e) * NN + nn] =
                    (_Float16)(acc[sub * 4 + t][r] + bv);
            }
        }
    }
}

// ---------------------------------------------------------------------------
// Flash attention, one wave per (b,h,16-query tile).
//   S^T = K * Q^T      (lane = query col, VGPR = key row)
//   two 16-key S^T tiles, exp()'d, pack exactly into one K=32 B-fragment
//   out^T += Vt * P    (A = Vt rows (e), B = P)
// Software-pipelined: next chunk's K-fragments and this chunk's V-fragments
// are issued before the S WMMAs; the softmax VALU work covers the V latency.
// Reference applies softmax FIRST then divides by sqrt(768) -> scores are raw,
// and 1/sqrt(768) folds into the final normalization.
// ---------------------------------------------------------------------------
__global__ void __launch_bounds__(128) attn_kernel(
        const _Float16* __restrict__ qb, const _Float16* __restrict__ kb,
        const _Float16* __restrict__ vt, _Float16* __restrict__ concat) {
    const int lane = threadIdx.x & 31;
    const int wid  = blockIdx.x * (blockDim.x >> 5) + (threadIdx.x >> 5);
    const int nQT  = NN / 16;                // 64
    const int qt   = wid % nQT;
    const int bh   = wid / nQT;
    if (bh >= BB * HH) return;
    const int b = bh / HH;
    const int h = bh % HH;

    const _Float16* qbase = qb + ((size_t)bh * NN + qt * 16) * EE;
    const _Float16* kbase = kb + (size_t)bh * NN * EE;
    const _Float16* vbase = vt + (size_t)bh * EE * NN;

    // Q fragments (B-operand of S^T): lane -> query col, K-dim = e (2 chunks)
    const v16h qf0 = load_frag16(qbase + 0,  EE, lane);
    const v16h qf1 = load_frag16(qbase + 32, EE, lane);

    v8f acc[4] = {};                          // out^T: 64(e) x 16(q)
    float m_run = -1e30f, l_run = 0.f;

    // preload chunk 0 K-fragments
    v16h kf0 = load_frag16(kbase + 0 * EE + 0,  EE, lane);
    v16h kf1 = load_frag16(kbase + 0 * EE + 32, EE, lane);
    v16h kf2 = load_frag16(kbase + 16 * EE + 0,  EE, lane);
    v16h kf3 = load_frag16(kbase + 16 * EE + 32, EE, lane);

    for (int c = 0; c < NN / 32; ++c) {
        const int key0 = c * 32;
        const int keyn = (c + 1 < NN / 32) ? (key0 + 32) : key0;  // last: cache hit

        // prefetch next chunk's K-fragments
        v16h nk0 = load_frag16(kbase + (size_t)keyn * EE + 0,  EE, lane);
        v16h nk1 = load_frag16(kbase + (size_t)keyn * EE + 32, EE, lane);
        v16h nk2 = load_frag16(kbase + (size_t)(keyn + 16) * EE + 0,  EE, lane);
        v16h nk3 = load_frag16(kbase + (size_t)(keyn + 16) * EE + 32, EE, lane);
        // prefetch this chunk's V-fragments (A-operand of out^T)
        v16h vf0 = load_frag16(vbase + (size_t)(0 * 16) * NN + key0, NN, lane);
        v16h vf1 = load_frag16(vbase + (size_t)(1 * 16) * NN + key0, NN, lane);
        v16h vf2 = load_frag16(vbase + (size_t)(2 * 16) * NN + key0, NN, lane);
        v16h vf3 = load_frag16(vbase + (size_t)(3 * 16) * NN + key0, NN, lane);

        // S^T tiles: keys key0..+15 and key0+16..+31 (resident fragments)
        v8f s0 = {};
        s0 = wmma_f16(kf0, qf0, s0);
        s0 = wmma_f16(kf1, qf1, s0);
        v8f s1 = {};
        s1 = wmma_f16(kf2, qf0, s1);
        s1 = wmma_f16(kf3, qf1, s1);

        // online softmax stats for this 32-key chunk (per query column):
        // per-lane reduce over 16 values, combine lane pair (l, l^16)
        float cmax = -1e30f;
#pragma unroll
        for (int r = 0; r < 8; ++r) {
            cmax = fmaxf(cmax, s0[r]);
            cmax = fmaxf(cmax, s1[r]);
        }
        cmax = fmaxf(cmax, __shfl_xor(cmax, 16, 32));
        const float m_new = fmaxf(m_run, cmax);
        const float scale = __expf(m_run - m_new);

        float p0[8], p1[8], psum = 0.f;
#pragma unroll
        for (int r = 0; r < 8; ++r) {
            p0[r] = __expf(s0[r] - m_new);
            p1[r] = __expf(s1[r] - m_new);
            psum += p0[r] + p1[r];
        }
        psum += __shfl_xor(psum, 16, 32);
        l_run = l_run * scale + psum;
        m_run = m_new;

        // pack probabilities into the K=32 B-fragment (pattern matches exactly)
        v16h pf;
#pragma unroll
        for (int r = 0; r < 8; ++r) {
            pf[r]     = (_Float16)p0[r];
            pf[r + 8] = (_Float16)p1[r];
        }

        // out^T accumulation with prefetched V-fragments
        {
            v8f a0 = acc[0], a1 = acc[1], a2 = acc[2], a3 = acc[3];
#pragma unroll
            for (int r = 0; r < 8; ++r) {
                a0[r] *= scale; a1[r] *= scale; a2[r] *= scale; a3[r] *= scale;
            }
            acc[0] = wmma_f16(vf0, pf, a0);
            acc[1] = wmma_f16(vf1, pf, a1);
            acc[2] = wmma_f16(vf2, pf, a2);
            acc[3] = wmma_f16(vf3, pf, a3);
        }

        kf0 = nk0; kf1 = nk1; kf2 = nk2; kf3 = nk3;
    }

    // normalize (softmax denom) and fold in the reference's 1/sqrt(768)
    const float inv = 1.0f / (l_run * 27.712812921102035f);
    const int n  = lane & 15;
    const int hi = lane >> 4;
    const int g  = b * NN + qt * 16 + n;                  // row in B*N
#pragma unroll
    for (int t = 0; t < 4; ++t) {
#pragma unroll
        for (int r = 0; r < 8; ++r) {
            const int e = t * 16 + r + 8 * hi;
            concat[(size_t)g * DD + h * EE + e] = (_Float16)(acc[t][r] * inv);
        }
    }
}

// ---------------------------------------------------------------------------
// Output projection: out = concat @ Wo^T + b  (f32 output)
// ---------------------------------------------------------------------------
__global__ void __launch_bounds__(128) proj_out_kernel(
        const _Float16* __restrict__ ch, const _Float16* __restrict__ wo,
        const float* __restrict__ bias, float* __restrict__ out) {
    const int lane = threadIdx.x & 31;
    const int wid  = blockIdx.x * (blockDim.x >> 5) + (threadIdx.x >> 5);
    const int nCG  = DD / 64;                // 12
    const int rt   = wid / nCG;
    const int cg   = wid % nCG;
    if (rt >= BN / 32) return;

    v8f acc[8] = {};
    gemm_32x64(ch + (size_t)rt * 32 * DD, wo + (size_t)cg * 64 * DD, lane, acc);

    const int n  = lane & 15;
    const int hi = lane >> 4;
#pragma unroll
    for (int t = 0; t < 4; ++t) {
        const int f  = cg * 64 + t * 16 + n;
        const float bv = bias[f];
#pragma unroll
        for (int sub = 0; sub < 2; ++sub) {
#pragma unroll
            for (int r = 0; r < 8; ++r) {
                const int g = rt * 32 + sub * 16 + r + 8 * hi;
                out[(size_t)g * DD + f] = acc[sub * 4 + t][r] + bv;
            }
        }
    }
}

// ---------------------------------------------------------------------------
extern "C" void kernel_launch(void* const* d_in, const int* in_sizes, int n_in,
                              void* d_out, int out_size, void* d_ws, size_t ws_size,
                              hipStream_t stream) {
    (void)in_sizes; (void)n_in; (void)out_size; (void)ws_size;
    const float* x     = (const float*)d_in[0];
    const float* wqk_w = (const float*)d_in[1];
    const float* wqk_b = (const float*)d_in[2];
    const float* wv_w  = (const float*)d_in[3];
    const float* wv_b  = (const float*)d_in[4];
    const float* wo_w  = (const float*)d_in[5];
    const float* wo_b  = (const float*)d_in[6];
    float* out = (float*)d_out;

    // workspace carve-up (256B aligned slabs), ~125 MB total
    char* ws = (char*)d_ws;
    size_t off = 0;
    auto alloc = [&](size_t bytes) -> char* {
        char* p = ws + off;
        off += (bytes + 255) & ~(size_t)255;
        return p;
    };
    _Float16* xh   = (_Float16*)alloc((size_t)BN * DD * 2);            // x f16
    _Float16* wqkh = (_Float16*)alloc((size_t)2 * DD * DD * 2);        // Wqk f16
    _Float16* wvh  = (_Float16*)alloc((size_t)DD * DD * 2);            // Wv f16
    _Float16* woh  = (_Float16*)alloc((size_t)DD * DD * 2);            // Wo f16
    _Float16* qbuf = (_Float16*)alloc((size_t)BB * HH * NN * EE * 2);  // Q (b,h,n,e)
    _Float16* kbuf = (_Float16*)alloc((size_t)BB * HH * NN * EE * 2);  // K (b,h,n,e)
    _Float16* vtb  = (_Float16*)alloc((size_t)BB * HH * EE * NN * 2);  // V^T (b,h,e,n)
    _Float16* cat  = (_Float16*)alloc((size_t)BN * DD * 2);            // concat f16

    // 1) convert inputs to f16
    cvt_f32_f16<<<4096, 256, 0, stream>>>(x,     xh,   BN * DD);
    cvt_f32_f16<<<1024, 256, 0, stream>>>(wqk_w, wqkh, 2 * DD * DD);
    cvt_f32_f16<<<512,  256, 0, stream>>>(wv_w,  wvh,  DD * DD);
    cvt_f32_f16<<<512,  256, 0, stream>>>(wo_w,  woh,  DD * DD);

    // 2) Q/K projection: (BN/32) row tiles x 24 column groups = 12288 waves
    proj_qk_kernel<<<(BN / 32) * 24 / 4, 128, 0, stream>>>(xh, wqkh, wqk_b, qbuf, kbuf);

    // 3) V projection: (BN/32) x 12 = 6144 waves
    proj_v_kernel<<<(BN / 32) * 12 / 4, 128, 0, stream>>>(xh, wvh, wv_b, vtb);

    // 4) flash attention: B*H*(N/16) = 12288 waves
    attn_kernel<<<BB * HH * (NN / 16) / 4, 128, 0, stream>>>(qbuf, kbuf, vtb, cat);

    // 5) output projection: 6144 waves, f32 out with bias
    proj_out_kernel<<<(BN / 32) * 12 / 4, 128, 0, stream>>>(cat, woh, wo_b, out);
}
